// SimpleVQ_13271448944641
// MI455X (gfx1250) — matrix operand order repaired and checked by reading, not compiled, not run on hardware
//
#include <hip/hip_runtime.h>
#include <hip/hip_bf16.h>
#include <math.h>

// Problem constants (from reference)
#define BB   32
#define LL   4096
#define DD   64
#define SS   512
#define ROWS (BB * LL)            // 131072 (H == 1)
#define PE_LAM 100000.0f
// tau^-0.5 = (sqrt(64))^-0.5 = 8^-0.5
#define TAU_SCALE 0.35355339059327373f

// padded row stride (in halves) for LDS matrices: 72*2B = 144B (16B aligned,
// 36-word stride -> 16 distinct banks for 16-lane column access)
#define CBSTRIDE 72

typedef __attribute__((ext_vector_type(16))) _Float16 v16h;
typedef __attribute__((ext_vector_type(8)))  _Float16 v8h;
typedef __attribute__((ext_vector_type(8)))  float    v8f;

// ---------------------------------------------------------------------------
// Kernel 1: build codebook. c[s] = rmsnorm(concat(sin,cos)) * tau^-0.5
// Outputs: c32 [512][64] f32, c16 [512][64] f16, cn2 [512] = ||c||^2
// ---------------------------------------------------------------------------
__global__ __launch_bounds__(512) void vq_codebook(float* __restrict__ c32,
                                                   float* __restrict__ cn2,
                                                   _Float16* __restrict__ c16) {
    int s = threadIdx.x;  // 0..511, single block
    if (s >= SS) return;
    float vals[DD];
    float ss = 0.f;
    float loglam = logf(PE_LAM);
#pragma unroll 4
    for (int j = 0; j < DD / 2; j++) {
        // inv_lam = lam^(-(2j)/D)
        float inv = expf(-((float)(2 * j) / (float)DD) * loglam);
        float pre = (float)s * inv;
        float sj = sinf(pre);
        float cj = cosf(pre);
        vals[j] = sj;
        vals[DD / 2 + j] = cj;
        ss += sj * sj + cj * cj;
    }
    float scale = rsqrtf(ss / (float)DD + 1e-6f) * TAU_SCALE;
    float s2 = 0.f;
#pragma unroll 4
    for (int k = 0; k < DD; k++) {
        float v = vals[k] * scale;
        c32[s * DD + k] = v;
        c16[s * DD + k] = (_Float16)v;
        s2 += v * v;
    }
    cn2[s] = s2;
}

// ---------------------------------------------------------------------------
// Kernel 2: fused distance GEMM (WMMA f16) + argmin + errs2 + codeword gather
// Block = 256 threads = 8 waves; each wave owns TWO 16-row tiles (32 rows),
// sharing B fragments across both (4 WMMAs per 4 ds_load_b128).
// Grid = ROWS / 256 = 512 blocks.
// Dynamic LDS layout (bytes):
//   [0)       f16 codebook, padded rows     512*72*2 = 73728
//   [73728)   ||c||^2                       512*4    =  2048
//   [75776)   per-wave A staging (padded)   8*32*72*2= 36864
//   [112640)  per-wave ||v||^2              8*32*4   =  1024
//   [113664)  per-wave errs2                8*32*4   =  1024
//   [114688)  per-wave z                    8*32*4   =  1024
//   [115712)  per-wave l_commit partials    8*4      =    32
// total 115744 B (< 320 KB WGP LDS; 2 blocks co-resident)
// ---------------------------------------------------------------------------
#define SHM_BYTES 115744

__global__ __launch_bounds__(256) void vq_main(const float* __restrict__ vecs,
                                               const float* __restrict__ mask,
                                               const float* __restrict__ c32,
                                               const float* __restrict__ cn2g,
                                               const _Float16* __restrict__ c16g,
                                               float* __restrict__ outq,
                                               int* __restrict__ outz,
                                               float* __restrict__ oute,
                                               float* __restrict__ partials) {
    extern __shared__ char smem[];
    _Float16* sCB  = (_Float16*)smem;                    // [512][72]
    float*    sCN2 = (float*)(smem + 73728);             // [512]
    _Float16* sA   = (_Float16*)(smem + 75776);          // [8][32][72]
    float*    sV2  = (float*)(smem + 112640);            // [8][32]
    float*    sE   = (float*)(smem + 113664);            // [8][32]
    int*      sZ   = (int*)(smem + 114688);              // [8][32]
    float*    sP   = (float*)(smem + 115712);            // [8]

    const int tid = threadIdx.x;
    const int w = tid >> 5;      // wave id 0..7
    const int t = tid & 31;      // lane id

    // ---- stage codebook into LDS (pad 64 -> 72 halves per row) ----
    {
        // 512 rows * 8 chunks of 8 halves (16B) = 4096 chunks
#pragma unroll
        for (int i = 0; i < 16; i++) {
            int idx = tid + 256 * i;
            int s = idx >> 3, chunk = idx & 7;
            *(uint4*)(sCB + s * CBSTRIDE + chunk * 8) =
                *(const uint4*)(c16g + s * DD + chunk * 8);
        }
        for (int i = tid; i < SS; i += 256) sCN2[i] = cn2g[i];
    }
    __syncthreads();

    const int rowBase = (blockIdx.x * 8 + w) * 32;   // first row of this wave's 32 rows

    // ---- load 32x64 f32 rows (one row per lane), fold -2 into f16 A operand ----
    {
        const float4* vsrc = (const float4*)(vecs + (size_t)(rowBase + t) * DD);
        _Float16* arow = sA + w * (32 * CBSTRIDE) + t * CBSTRIDE;
        float ss = 0.f;
#pragma unroll
        for (int i = 0; i < 8; i++) {           // two float4 -> one 16B LDS store
            float4 va = vsrc[2 * i];
            float4 vb = vsrc[2 * i + 1];
            ss += va.x * va.x + va.y * va.y + va.z * va.z + va.w * va.w;
            ss += vb.x * vb.x + vb.y * vb.y + vb.z * vb.z + vb.w * vb.w;
            v8h p;
            p[0] = (_Float16)(-2.f * va.x); p[1] = (_Float16)(-2.f * va.y);
            p[2] = (_Float16)(-2.f * va.z); p[3] = (_Float16)(-2.f * va.w);
            p[4] = (_Float16)(-2.f * vb.x); p[5] = (_Float16)(-2.f * vb.y);
            p[6] = (_Float16)(-2.f * vb.z); p[7] = (_Float16)(-2.f * vb.w);
            *(v8h*)(arow + 8 * i) = p;
        }
        sV2[w * 32 + t] = ss;
    }

    // ---- build A fragments (ISA 16-bit A 16x32 layout) for both M-tiles ----
    // lanes 0-15: halves 0..7 -> K=c..c+7, halves 8..15 -> K=c+16..c+23
    // lanes 16-31: same with +8 offset
    const int M = t & 15;
    const int koff = (t >> 4) << 3;                    // 0 or 8
    const _Float16* aM0 = sA + w * (32 * CBSTRIDE) + M * CBSTRIDE;
    const _Float16* aM1 = aM0 + 16 * CBSTRIDE;
    v16h a0, a1, a2, a3;
    {
        v8h lo0 = *(const v8h*)(aM0 + koff);
        v8h hi0 = *(const v8h*)(aM0 + koff + 16);
        v8h lo1 = *(const v8h*)(aM0 + 32 + koff);
        v8h hi1 = *(const v8h*)(aM0 + 32 + koff + 16);
        v8h lo2 = *(const v8h*)(aM1 + koff);
        v8h hi2 = *(const v8h*)(aM1 + koff + 16);
        v8h lo3 = *(const v8h*)(aM1 + 32 + koff);
        v8h hi3 = *(const v8h*)(aM1 + 32 + koff + 16);
#pragma unroll
        for (int i = 0; i < 8; i++) {
            a0[i] = lo0[i]; a0[i + 8] = hi0[i];
            a1[i] = lo1[i]; a1[i + 8] = hi1[i];
            a2[i] = lo2[i]; a2[i + 8] = hi2[i];
            a3[i] = lo3[i]; a3[i + 8] = hi3[i];
        }
    }

    // ---- sweep all 512 codewords in 16-column WMMA tiles, B shared by both M-tiles
    // B layout (32x16 f16): lanes 0-15 hold K=0..7,16..23, lanes 16-31 hold K=8..15,24..31, N=lane&15
    const int nlane = t & 15;
    const int kb = (t >> 4) << 3;                      // 0 or 8
    float best[16];
    int   bidx[16];
#pragma unroll
    for (int r = 0; r < 16; r++) { best[r] = 3.4e38f; bidx[r] = 0; }

    for (int n0 = 0; n0 < SS; n0 += 16) {
        const int ncol = n0 + nlane;
        const float cn2v = sCN2[ncol];
        const _Float16* brow = sCB + ncol * CBSTRIDE + kb;
        v8h q0 = *(const v8h*)(brow);
        v8h q1 = *(const v8h*)(brow + 16);
        v8h q2 = *(const v8h*)(brow + 32);
        v8h q3 = *(const v8h*)(brow + 48);
        v16h b0, b1;
#pragma unroll
        for (int i = 0; i < 8; i++) {
            b0[i] = q0[i]; b0[i + 8] = q1[i];
            b1[i] = q2[i]; b1[i + 8] = q3[i];
        }
        v8f zc = {};
        // acc = (-2v)·c ; ||c||^2 added as a scalar in the compare phase
        v8f acc0 = __builtin_amdgcn_wmma_f32_16x16x32_f16(false, a0, false, b0,
                                                          (short)0, zc, false, false);
        v8f acc1 = __builtin_amdgcn_wmma_f32_16x16x32_f16(false, a2, false, b0,
                                                          (short)0, zc, false, false);
        acc0 = __builtin_amdgcn_wmma_f32_16x16x32_f16(false, a1, false, b1,
                                                      (short)0, acc0, false, false);
        acc1 = __builtin_amdgcn_wmma_f32_16x16x32_f16(false, a3, false, b1,
                                                      (short)0, acc1, false, false);
#pragma unroll
        for (int r = 0; r < 8; r++) {
            float sc0 = acc0[r] + cn2v;
            float sc1 = acc1[r] + cn2v;
            bool l0 = sc0 < best[r];
            bool l1 = sc1 < best[r + 8];
            best[r]     = l0 ? sc0  : best[r];
            bidx[r]     = l0 ? ncol : bidx[r];
            best[r + 8] = l1 ? sc1  : best[r + 8];
            bidx[r + 8] = l1 ? ncol : bidx[r + 8];
        }
    }

    // ---- branchless argmin reduction across the 16 lanes of each half ----
#pragma unroll
    for (int off = 1; off < 16; off <<= 1) {
#pragma unroll
        for (int r = 0; r < 16; r++) {
            float ob = __shfl_xor(best[r], off);
            int   oi = __shfl_xor(bidx[r], off);
            int take = (int)(ob < best[r]) |
                       ((int)(ob == best[r]) & (int)(oi < bidx[r]));
            best[r] = take ? ob : best[r];
            bidx[r] = take ? oi : bidx[r];
        }
    }

    // ---- lanes 0 and 16 own rows {M-half} of both tiles ----
    if ((t & 15) == 0) {
        const int half = t >> 4;
#pragma unroll
        for (int r = 0; r < 16; r++) {
            const int rl = ((r >> 3) << 4) + (r & 7) + half * 8;  // local row 0..31
            const float v2 = sV2[w * 32 + rl];
            const float dist = v2 + best[r];           // ||v||^2 - 2v.c + ||c||^2
            const float err = fmaxf(dist, 0.f);        // relu(min)
            const int row = rowBase + rl;
            outz[row] = bidx[r];
            oute[row] = err;
            sE[w * 32 + rl] = err;
            sZ[w * 32 + rl] = bidx[r];
        }
    }

    // ---- wave-local masked l_commit partial (fixed order -> deterministic) ----
    if (t == 0) {
        float sum = 0.f;
#pragma unroll
        for (int i = 0; i < 32; i++) sum += sE[w * 32 + i] * mask[rowBase + i];
        sP[w] = sum;
    }

    // ---- quantized = exact f32 codeword gather: each lane writes one row ----
    {
        const int z = sZ[w * 32 + t];
        const float4* qs = (const float4*)(c32 + (size_t)z * DD);
        float4* qd = (float4*)(outq + (size_t)(rowBase + t) * DD);
#pragma unroll
        for (int i = 0; i < 16; i++) qd[i] = qs[i];
    }

    __syncthreads();
    if (tid == 0) {
        float s = 0.f;
#pragma unroll
        for (int i = 0; i < 8; i++) s += sP[i];
        partials[blockIdx.x] = s;
    }
}

// ---------------------------------------------------------------------------
// Kernel 3: deterministic final reduction of per-block partials -> l_commit
// ---------------------------------------------------------------------------
__global__ __launch_bounds__(256) void vq_lcreduce(const float* __restrict__ partials,
                                                   float* __restrict__ out_lc, int n) {
    __shared__ float sm[256];
    float s = 0.f;
    for (int i = threadIdx.x; i < n; i += 256) s += partials[i];
    sm[threadIdx.x] = s;
    __syncthreads();
#pragma unroll
    for (int stride = 128; stride > 0; stride >>= 1) {
        if (threadIdx.x < stride) sm[threadIdx.x] += sm[threadIdx.x + stride];
        __syncthreads();
    }
    if (threadIdx.x == 0) out_lc[0] = sm[0] * (1.0f / (float)ROWS);
}

// ---------------------------------------------------------------------------
// Host entry
// ---------------------------------------------------------------------------
extern "C" void kernel_launch(void* const* d_in, const int* in_sizes, int n_in,
                              void* d_out, int out_size, void* d_ws, size_t ws_size,
                              hipStream_t stream) {
    (void)in_sizes; (void)n_in; (void)out_size; (void)ws_size;
    const float* vecs = (const float*)d_in[0];   // [B,H,L,D] f32
    const float* mask = (const float*)d_in[1];   // [B,L] f32

    // Output: (quantized [B,H,L,D] f32, z [B,H,L] i32, l_commit f32, errs2 [B,H,L] f32)
    float* out   = (float*)d_out;
    float* outq  = out;                          // ROWS*64
    int*   outz  = (int*)(out + (size_t)ROWS * DD);
    float* outlc = out + (size_t)ROWS * DD + ROWS;
    float* oute  = outlc + 1;

    // Workspace: c32 | cn2 | c16 | block partials
    float*    c32 = (float*)d_ws;                              // 512*64 f32
    float*    cn2 = c32 + SS * DD;                             // 512 f32
    _Float16* c16 = (_Float16*)(cn2 + SS);                     // 512*64 f16
    float*    partials = (float*)((char*)d_ws +
                         (size_t)SS * DD * 4 + SS * 4 + (size_t)SS * DD * 2); // 512 f32

    vq_codebook<<<1, 512, 0, stream>>>(c32, cn2, c16);

    const int nblocks = ROWS / (32 * 8);   // 512
    vq_main<<<nblocks, 256, SHM_BYTES, stream>>>(vecs, mask, c32, cn2, c16,
                                                 outq, outz, oute, partials);

    vq_lcreduce<<<1, 256, 0, stream>>>(partials, outlc, nblocks);
}
